// SelfAttention_47863115547190
// MI455X (gfx1250) — matrix operand order repaired and checked by reading, not compiled, not run on hardware
//
#include <hip/hip_runtime.h>

// ---------------------------------------------------------------------------
// MI455X (gfx1250) flash self-attention.
//   K1: QKV 1x1-conv projections via v_wmma_f32_16x16x32_bf16 (f32 -> bf16)
//   K2: flash attention (online softmax), 20 WMMA per 32-j chunk, K/V chunks
//       double-buffered into LDS with global_load_async_to_lds_b128.
// ---------------------------------------------------------------------------

typedef __bf16 bf16_t;
typedef bf16_t v16bf __attribute__((ext_vector_type(16)));
typedef bf16_t v2bf  __attribute__((ext_vector_type(2)));
typedef float  v8f   __attribute__((ext_vector_type(8)));
typedef int    v4i_t __attribute__((ext_vector_type(4)));

#define B_   8
#define C_   256
#define C4_  64
#define N_   4096
#define QK_ELEMS ((size_t)B_ * (size_t)N_ * (size_t)C4_)   // 2,097,152 bf16

union BF16Frag {
  v16bf        v;
  unsigned short u[16];
  unsigned int   w[8];
};

// Hardware f32 -> bf16 conversions (RTNE on CDNA5).
__device__ inline unsigned short f2bf(float f) {
  union { bf16_t b; unsigned short u; } t;
  t.b = (bf16_t)f;
  return t.u;
}

__device__ inline unsigned int pk2bf(float lo, float hi) {
#if defined(__AMDGCN__) && __has_builtin(__builtin_amdgcn_cvt_pk_bf16_f32)
  union { v2bf v; unsigned int w; } t;
  t.v = __builtin_amdgcn_cvt_pk_bf16_f32(lo, hi);
  return t.w;
#else
  union { bf16_t b[2]; unsigned int w; } t;
  t.b[0] = (bf16_t)lo; t.b[1] = (bf16_t)hi;
  return t.w;
#endif
}

#if defined(__AMDGCN__) && __has_builtin(__builtin_amdgcn_global_load_async_to_lds_b128)
#define USE_ASYNC 1
#else
#define USE_ASYNC 0
#endif

#if defined(__AMDGCN__)
typedef __attribute__((address_space(1))) v4i_t* gv4i_ptr;
typedef __attribute__((address_space(3))) v4i_t* lv4i_ptr;
#endif

__device__ inline void cp16(const void* g, void* l) {
#if USE_ASYNC
  __builtin_amdgcn_global_load_async_to_lds_b128((gv4i_ptr)g, (lv4i_ptr)l, 0, 0);
#else
  *(uint4*)l = *(const uint4*)g;
#endif
}

__device__ inline void cp_wait0() {
#if USE_ASYNC
#if __has_builtin(__builtin_amdgcn_s_wait_asynccnt)
  __builtin_amdgcn_s_wait_asynccnt(0);
#else
  asm volatile("s_wait_asynccnt 0" ::: "memory");
#endif
#endif
}

// ---------------------------------------------------------------------------
// Kernel 1: q = Wq x + bq, k = Wk ref + bk, v = Wv x + bv
// q,k stored pixel-major [B][N][64]; v stored channel-major [B][C][N] (bf16).
// Block = 16 pixels of one batch image; 8 waves x 3 output tiles each.
// ---------------------------------------------------------------------------
__global__ __launch_bounds__(256, 1) void qkv_proj_kernel(
    const float* __restrict__ x,  const float* __restrict__ ref,
    const float* __restrict__ Wq, const float* __restrict__ bq,
    const float* __restrict__ Wk, const float* __restrict__ bk,
    const float* __restrict__ Wv, const float* __restrict__ bv,
    unsigned short* __restrict__ qt, unsigned short* __restrict__ kt,
    unsigned short* __restrict__ vt)
{
  const int b  = blockIdx.y;
  const int p0 = blockIdx.x * 16;

  __shared__ __align__(16) unsigned short lx[16][C_];   // x slab, bf16
  __shared__ __align__(16) unsigned short lr[16][C_];   // ref slab, bf16

  for (int e = threadIdx.x; e < 16 * C_; e += 256) {
    const int c  = e >> 4;
    const int pl = e & 15;
    const size_t gi = ((size_t)(b * C_ + c)) * N_ + p0 + pl;
    lx[pl][c] = f2bf(x[gi]);
    lr[pl][c] = f2bf(ref[gi]);
  }
  __syncthreads();

  const int wave = threadIdx.x >> 5;
  const int lane = threadIdx.x & 31;
  const int m  = lane & 15;   // C/D column, A row
  const int hi = lane >> 4;   // lane-half select

  for (int tt = 0; tt < 3; ++tt) {
    const int tix = wave * 3 + tt;            // 0..23: 4 q-tiles, 4 k, 16 v
    const unsigned short (*src)[C_];
    const float* W; const float* bias;
    int obase; bool isV;
    if (tix < 4)      { src = lx; W = Wq; bias = bq; obase = tix * 16;       isV = false; }
    else if (tix < 8) { src = lr; W = Wk; bias = bk; obase = (tix - 4) * 16; isV = false; }
    else              { src = lx; W = Wv; bias = bv; obase = (tix - 8) * 16; isV = true;  }

    const int o = obase + m;                  // this lane's output channel
    v8f acc = {};
#pragma unroll
    for (int kk = 0; kk < C_; kk += 32) {
      BF16Frag a, bm;
#pragma unroll
      for (int vg = 0; vg < 8; ++vg) {
        // A (16x32 bf16): lanes 0-15 K=0..15; lanes 16-31 K=16..31
        const int ka = kk + hi * 16 + vg * 2;
        a.w[vg] = *(const unsigned int*)&src[m][ka];
        // B (32x16): lane holds column o; K pairs 2vg,2vg+1 (+16 for hi lanes)
        const int kb = kk + hi * 16 + vg * 2;
        bm.w[vg] = pk2bf(W[(size_t)o * C_ + kb], W[(size_t)o * C_ + kb + 1]);
      }
      acc = __builtin_amdgcn_wmma_f32_16x16x32_bf16(false, a.v, false, bm.v,
                                                    (short)0, acc, false, false);
    }
    const float bb = bias[o];
#pragma unroll
    for (int r = 0; r < 8; ++r) {             // D: lane-half hi rows hi*8+r
      const float val = acc[r] + bb;
      const int p = p0 + hi * 8 + r;
      if (!isV) {
        unsigned short* dst = (tix < 4) ? qt : kt;
        dst[((size_t)b * N_ + p) * C4_ + o] = f2bf(val);
      } else {
        vt[((size_t)(b * C_ + o)) * N_ + p] = f2bf(val);
      }
    }
  }
}

// ---------------------------------------------------------------------------
// Async-stage one 32-j chunk: K rows (contiguous 4KB) and V rows (channel-
// major, 64B each, transposed placement into svb[c][32]).
// ---------------------------------------------------------------------------
__device__ inline void issue_chunk(const unsigned short* __restrict__ kt,
                                   const unsigned short* __restrict__ vt,
                                   unsigned short* skb, unsigned short* svb,
                                   int b, int j0, int lane)
{
  const unsigned short* kg = kt + ((size_t)b * N_ + j0) * C4_;
#pragma unroll
  for (int off = 0; off < 32 * C4_; off += 32 * 8)          // 8 async b128 ops
    cp16(kg + off + lane * 8, skb + off + lane * 8);

  const int crow = lane >> 2;        // 8 channel rows per instruction
  const int seg  = (lane & 3) * 8;   // 16B segment within the 64B row
  for (int cb = 0; cb < C_; cb += 8) {
    const int c = cb + crow;
    cp16(vt + ((size_t)(b * C_ + c)) * N_ + j0 + seg, svb + c * 32 + seg);
  }
}

// ---------------------------------------------------------------------------
// Kernel 2: flash attention for one 16-row i-tile. One wave32 per block.
// O[16 c-tiles] kept in 128 f32 VGPRs; 20 WMMA per 32-column j chunk.
// ---------------------------------------------------------------------------
__global__ __launch_bounds__(32, 1) void flash_attn_kernel(
    const float* __restrict__ x, const float* __restrict__ gamma,
    const unsigned short* __restrict__ qt, const unsigned short* __restrict__ kt,
    const unsigned short* __restrict__ vt, float* __restrict__ out)
{
  const int b  = blockIdx.y;
  const int i0 = blockIdx.x * 16;
  const int lane = threadIdx.x & 31;
  const int m  = lane & 15;
  const int hi = lane >> 4;

  __shared__ __align__(16) unsigned short sk[2][32 * C4_];  // K chunk, [j][64]
  __shared__ __align__(16) unsigned short sv[2][C_ * 32];   // V chunk, [c][32]
  __shared__ __align__(16) unsigned short sp[16 * 32];      // P transpose buf

  // Q fragments for the whole i-tile, kept in registers (2 k-steps of 32)
  BF16Frag qa[2];
#pragma unroll
  for (int ks = 0; ks < 2; ++ks)
#pragma unroll
    for (int vg = 0; vg < 8; ++vg) {
      const int k = ks * 32 + hi * 16 + vg * 2;
      qa[ks].w[vg] = *(const unsigned int*)&qt[((size_t)b * N_ + i0 + m) * C4_ + k];
    }

  v8f O[16];
#pragma unroll
  for (int ct = 0; ct < 16; ++ct) O[ct] = (v8f){};
  float mrow[8], lrow[8];
#pragma unroll
  for (int r = 0; r < 8; ++r) { mrow[r] = -3.0e38f; lrow[r] = 0.0f; }
  const float scale = 0.125f;  // 1/sqrt(64)

  issue_chunk(kt, vt, sk[0], sv[0], b, 0, lane);

  for (int ch = 0; ch < N_ / 32; ++ch) {
    const int cur = ch & 1;
    cp_wait0();
    __syncthreads();
    if (ch + 1 < N_ / 32)
      issue_chunk(kt, vt, sk[cur ^ 1], sv[cur ^ 1], b, (ch + 1) * 32, lane);

    const unsigned short* skb = sk[cur];
    const unsigned short* svb = sv[cur];

    // ---- S = Q * K^T : two 16x16 sub-tiles (j columns 0..15 / 16..31)
    v8f s0 = {}, s1 = {};
#pragma unroll
    for (int ks = 0; ks < 2; ++ks) {
      BF16Frag b0, b1;
#pragma unroll
      for (int vg = 0; vg < 8; ++vg) {
        const int c = ks * 32 + hi * 16 + vg * 2;
        b0.w[vg] = *(const unsigned int*)&skb[(0 * 16 + m) * C4_ + c];
        b1.w[vg] = *(const unsigned int*)&skb[(1 * 16 + m) * C4_ + c];
      }
      s0 = __builtin_amdgcn_wmma_f32_16x16x32_bf16(false, qa[ks].v, false, b0.v, (short)0, s0, false, false);
      s1 = __builtin_amdgcn_wmma_f32_16x16x32_bf16(false, qa[ks].v, false, b1.v, (short)0, s1, false, false);
    }

    // ---- online softmax: rows hi*8+r live on this lane-half (16 lanes/cols)
    float pmax[8];
#pragma unroll
    for (int r = 0; r < 8; ++r) pmax[r] = fmaxf(s0[r], s1[r]) * scale;
#pragma unroll
    for (int d = 8; d >= 1; d >>= 1)
#pragma unroll
      for (int r = 0; r < 8; ++r) pmax[r] = fmaxf(pmax[r], __shfl_xor(pmax[r], d, 32));

    float alpha[8], psum[8];
#pragma unroll
    for (int r = 0; r < 8; ++r) {
      const float mnew = fmaxf(mrow[r], pmax[r]);
      alpha[r] = __expf(mrow[r] - mnew);
      mrow[r] = mnew;
      const float p0 = __expf(s0[r] * scale - mnew);
      const float p1 = __expf(s1[r] * scale - mnew);
      sp[(hi * 8 + r) * 32 + m]      = f2bf(p0);
      sp[(hi * 8 + r) * 32 + 16 + m] = f2bf(p1);
      psum[r] = p0 + p1;
    }
#pragma unroll
    for (int d = 8; d >= 1; d >>= 1)
#pragma unroll
      for (int r = 0; r < 8; ++r) psum[r] += __shfl_xor(psum[r], d, 32);
#pragma unroll
    for (int r = 0; r < 8; ++r) lrow[r] = lrow[r] * alpha[r] + psum[r];

#pragma unroll
    for (int ct = 0; ct < 16; ++ct)
#pragma unroll
      for (int r = 0; r < 8; ++r) O[ct][r] *= alpha[r];

    __syncthreads();

    // ---- A-fragment of P (16 rows x 32 j) from the LDS transpose buffer
    BF16Frag pa;
#pragma unroll
    for (int vg = 0; vg < 8; ++vg) {
      const int k = hi * 16 + vg * 2;
      pa.w[vg] = *(const unsigned int*)&sp[m * 32 + k];
    }

    // ---- O += P * V : 16 channel tiles
#pragma unroll
    for (int ct = 0; ct < 16; ++ct) {
      BF16Frag bvf;
#pragma unroll
      for (int vg = 0; vg < 8; ++vg) {
        const int j = hi * 16 + vg * 2;             // K pair = consecutive j
        bvf.w[vg] = *(const unsigned int*)&svb[(ct * 16 + m) * 32 + j];
      }
      O[ct] = __builtin_amdgcn_wmma_f32_16x16x32_bf16(false, pa.v, false, bvf.v, (short)0, O[ct], false, false);
    }
    __syncthreads();
  }

  // ---- epilogue: out = gamma * (O / l) + x
  const float g = gamma[0];
  float inv[8];
#pragma unroll
  for (int r = 0; r < 8; ++r) inv[r] = 1.0f / lrow[r];
#pragma unroll
  for (int ct = 0; ct < 16; ++ct) {
    const int c = ct * 16 + m;
#pragma unroll
    for (int r = 0; r < 8; ++r) {
      const int i = i0 + hi * 8 + r;
      const size_t idx = ((size_t)(b * C_ + c)) * N_ + i;
      out[idx] = g * (O[ct][r] * inv[r]) + x[idx];
    }
  }
}

// ---------------------------------------------------------------------------
extern "C" void kernel_launch(void* const* d_in, const int* in_sizes, int n_in,
                              void* d_out, int out_size, void* d_ws, size_t ws_size,
                              hipStream_t stream)
{
  (void)in_sizes; (void)n_in; (void)out_size; (void)ws_size;
  const float* x   = (const float*)d_in[0];
  const float* ref = (const float*)d_in[1];
  const float* Wq  = (const float*)d_in[2];
  const float* bq  = (const float*)d_in[3];
  const float* Wk  = (const float*)d_in[4];
  const float* bk  = (const float*)d_in[5];
  const float* Wv  = (const float*)d_in[6];
  const float* bv  = (const float*)d_in[7];
  const float* gamma = (const float*)d_in[8];
  float* out = (float*)d_out;

  unsigned short* qt = (unsigned short*)d_ws;       // [B][N][64] bf16
  unsigned short* kt = qt + QK_ELEMS;               // [B][N][64] bf16
  unsigned short* vt = kt + QK_ELEMS;               // [B][C][N]  bf16

  dim3 grid(N_ / 16, B_);
  qkv_proj_kernel<<<grid, 256, 0, stream>>>(x, ref, Wq, bq, Wk, bk, Wv, bv, qt, kt, vt);
  flash_attn_kernel<<<grid, 32, 0, stream>>>(x, gamma, qt, kt, vt, out);
}
